// ERD_37881611550853
// MI455X (gfx1250) — matrix operand order repaired and checked by reading, not compiled, not run on hardware
//
#include <hip/hip_runtime.h>

// ---------------------------------------------------------------------------
// ERD motion model decoder on MI455X (gfx1250), f16 WMMA with f32 accumulate.
// Encoder is dead code in the reference (state deleted) -> skipped entirely.
// Round 4: __builtin_amdgcn_sched_barrier(0) pins all 20 fragment ds_loads of
// a K-tile ahead of the 8 WMMAs, defeating the scheduler's just-in-time load
// rematerialization so DScnt waits become partial and WMMAs pipeline.
// ---------------------------------------------------------------------------

typedef __attribute__((ext_vector_type(16))) _Float16 v16h;
typedef __attribute__((ext_vector_type(8)))  _Float16 v8h;
typedef __attribute__((ext_vector_type(8)))  float    v8f;

#define ERD_B      1024
#define ERD_TOUT   25
#define ERD_DOUT   96

#define LDS_STRIDE 72   // halves per row (64 data + 8 pad) -> 144B, 16B aligned

// ---------------------------------------------------------------------------
// GEMM: C[M,N] = act(A[M,K] @ W[N,K]^T + bias)
//   A, W are f16 row-major; K multiple of 64; M mult of 64, N mult of 128.
// MODE 0: f32 store, + bias, no activation            (gates pass 1)
// MODE 1: f32 accumulate (C += A.W^T), no bias        (gates pass 2: h @ Whh^T)
// MODE 2: f16 store, + bias, relu                     (fc layers)
// MODE 3: like 2, additionally store f32 into out2 for cols < out2_cols (fcout)
// Block = 256 threads (8 wave32s). Block tile 64x128, BK=64, ping-pong LDS,
// one barrier per K-step; wave computes 16x64 = 8 WMMAs per K-step.
// ---------------------------------------------------------------------------
template <int MODE>
__global__ __launch_bounds__(256) void erd_gemm_wmma(
    const _Float16* __restrict__ A, const _Float16* __restrict__ W,
    const float* __restrict__ bias,
    float* __restrict__ Cf, _Float16* __restrict__ Ch,
    float* __restrict__ out2, int out2_cols, int out2_stride,
    int N, int K)
{
  __shared__ __align__(16) _Float16 AsBuf[2][64 * LDS_STRIDE];
  __shared__ __align__(16) _Float16 WsBuf[2][128 * LDS_STRIDE];

  const int tid  = threadIdx.x;
  const int lane = tid & 31;
  const int wave = tid >> 5;
  const int wy   = wave >> 1;        // 0..3 : 16-row strip
  const int wx   = wave & 1;         // 0..1 : 64-col half
  const int lr   = lane & 15;
  const int hi   = lane >> 4;        // 0/1 : half-wave K split (ISA 7.12.2)

  const int m0 = blockIdx.y * 64;
  const int n0 = blockIdx.x * 128;

  // global->reg staging assignments
  const int arow = tid >> 2;             // 0..63
  const int acol = (tid & 3) * 16;       // 0,16,32,48
  const int wrow = tid >> 1;             // 0..127
  const int wcol = (tid & 1) * 32;       // 0,32

  const _Float16* Ag = A + (size_t)(m0 + arow) * K + acol;
  const _Float16* Wg = W + (size_t)(n0 + wrow) * K + wcol;

  v8h aR0, aR1, wR0, wR1, wR2, wR3;
  auto load_tile = [&](int k0) {
    aR0 = *(const v8h*)(Ag + k0);
    aR1 = *(const v8h*)(Ag + k0 + 8);
    wR0 = *(const v8h*)(Wg + k0);
    wR1 = *(const v8h*)(Wg + k0 + 8);
    wR2 = *(const v8h*)(Wg + k0 + 16);
    wR3 = *(const v8h*)(Wg + k0 + 24);
  };
  auto store_tile = [&](int buf) {
    _Float16* as = &AsBuf[buf][arow * LDS_STRIDE + acol];
    _Float16* ws = &WsBuf[buf][wrow * LDS_STRIDE + wcol];
    *(v8h*)(as)     = aR0;
    *(v8h*)(as + 8) = aR1;
    *(v8h*)(ws)      = wR0;
    *(v8h*)(ws + 8)  = wR1;
    *(v8h*)(ws + 16) = wR2;
    *(v8h*)(ws + 24) = wR3;
  };

  v8f acc[4] = {v8f{}, v8f{}, v8f{}, v8f{}};

  load_tile(0);
  store_tile(0);
  int cur = 0;

  for (int k0 = 0; k0 < K; k0 += 64) {
    __syncthreads();                       // LDS[cur] ready; LDS[cur^1] free
    const bool more = (k0 + 64) < K;
    if (more) load_tile(k0 + 64);          // prefetch next tile (hidden by WMMA)

    const _Float16* as  = &AsBuf[cur][(wy * 16 + lr) * LDS_STRIDE];
    const _Float16* wsb = &WsBuf[cur][(wx * 64 + lr) * LDS_STRIDE];

    // ---- preload ALL fragments of this K-tile into distinct registers ----
    v16h afrag[2];
    v16h bfrag[8];
#pragma unroll
    for (int s = 0; s < 2; ++s) {
      v8h alo = *(const v8h*)(as + s * 32 + hi * 8);
      v8h ahi = *(const v8h*)(as + s * 32 + hi * 8 + 16);
      afrag[s] = __builtin_shufflevector(alo, ahi,
                   0,1,2,3,4,5,6,7,8,9,10,11,12,13,14,15);
#pragma unroll
      for (int j = 0; j < 4; ++j) {
        const _Float16* q = wsb + j * 16 * LDS_STRIDE + s * 32 + hi * 8;
        v8h blo = *(const v8h*)(q);
        v8h bhi = *(const v8h*)(q + 16);
        bfrag[s * 4 + j] = __builtin_shufflevector(blo, bhi,
                             0,1,2,3,4,5,6,7,8,9,10,11,12,13,14,15);
      }
    }

    // Pin the 20 ds_loads above this point: nothing may cross, so the loads
    // cannot be rematerialized just-in-time in front of each WMMA. DS returns
    // are in-order, so the waitcnt pass emits partial s_wait_dscnt values.
    __builtin_amdgcn_sched_barrier(0);

    // ---- 8 back-to-back WMMAs (4 independent accumulator chains) ----
#pragma unroll
    for (int s = 0; s < 2; ++s) {
#pragma unroll
      for (int j = 0; j < 4; ++j) {
        acc[j] = __builtin_amdgcn_wmma_f32_16x16x32_f16(
            false, afrag[s], false, bfrag[s * 4 + j], (short)0, acc[j],
            false, false);
      }
    }

    // Keep the ping-pong LDS stores (and their staged loadcnt waits) out of
    // the WMMA issue window.
    __builtin_amdgcn_sched_barrier(0);

    if (more) store_tile(cur ^ 1);
    cur ^= 1;
  }

  // C/D layout: VGPR v -> row (v + hi*8), col = lr (per 16x16 f32 table).
  const int rbase = m0 + wy * 16 + hi * 8;
  int   cc[4];
  float bb[4] = {0.f, 0.f, 0.f, 0.f};
#pragma unroll
  for (int j = 0; j < 4; ++j) {
    cc[j] = n0 + wx * 64 + j * 16 + lr;
    if (MODE != 1) bb[j] = bias[cc[j]];
  }

#pragma unroll
  for (int v = 0; v < 8; ++v) {
    const size_t r = (size_t)(rbase + v);
#pragma unroll
    for (int j = 0; j < 4; ++j) {
      const size_t i0 = r * (size_t)N + cc[j];
      float x = acc[j][v];
      if (MODE == 0) {
        Cf[i0] = x + bb[j];
      } else if (MODE == 1) {
        Cf[i0] += x;
      } else {
        x = fmaxf(x + bb[j], 0.f);
        Ch[i0] = (_Float16)x;
        if (MODE == 3) {
          if (cc[j] < out2_cols) out2[r * (size_t)out2_stride + cc[j]] = x;
        }
      }
    }
  }
}

// ---------------------------------------------------------------------------
// LSTM pointwise cell. gates g: [B,4096] f32, gate order i,f,g,o at column
// offsets 0/1000/2000/3000. Updates c (f32 [B,1024]) and h (f16 [B,1024],
// pad cols forced to zero since h feeds the next GEMM as a K=1024 operand).
// ---------------------------------------------------------------------------
__device__ __forceinline__ float erd_sigmoid(float x) {
  return 1.f / (1.f + __expf(-x));
}

__global__ __launch_bounds__(256) void erd_lstm_pointwise(
    const float* __restrict__ g, float* __restrict__ c,
    _Float16* __restrict__ h)
{
  const int idx = blockIdx.x * blockDim.x + threadIdx.x;  // B*1024
  const int r = idx >> 10;
  const int j = idx & 1023;
  if (j < 1000) {
    const float* gr = g + (size_t)r * 4096;
    const float ig = erd_sigmoid(gr[j]);
    const float fg = erd_sigmoid(gr[1000 + j]);
    const float gg = tanhf(gr[2000 + j]);
    const float og = erd_sigmoid(gr[3000 + j]);
    const size_t ci = (size_t)r * 1024 + j;
    const float c2 = fg * c[ci] + ig * gg;
    c[ci] = c2;
    h[ci] = (_Float16)(og * tanhf(c2));
  } else {
    h[(size_t)r * 1024 + j] = (_Float16)0.f;
  }
}

// ---------------------------------------------------------------------------
// Setup kernels: f32 -> padded f16 weights, padded (summed) biases, padded p.
// ---------------------------------------------------------------------------
__global__ __launch_bounds__(256) void erd_cvt_weight_pad(
    const float* __restrict__ src, _Float16* __restrict__ dst,
    int N, int K, int Npad, int Kpad)
{
  const int idx = blockIdx.x * blockDim.x + threadIdx.x;
  if (idx >= Npad * Kpad) return;
  const int n = idx / Kpad;
  const int k = idx - n * Kpad;
  dst[idx] = (n < N && k < K) ? (_Float16)src[(size_t)n * K + k] : (_Float16)0.f;
}

__global__ __launch_bounds__(256) void erd_pad_bias(
    const float* __restrict__ b1, const float* __restrict__ b2,
    float* __restrict__ dst, int N, int Npad)
{
  const int idx = blockIdx.x * blockDim.x + threadIdx.x;
  if (idx >= Npad) return;
  float v = 0.f;
  if (idx < N) {
    v = b1[idx];
    if (b2) v += b2[idx];
  }
  dst[idx] = v;
}

__global__ __launch_bounds__(256) void erd_cvt_p(
    const float* __restrict__ p, _Float16* __restrict__ prev)
{
  const int idx = blockIdx.x * blockDim.x + threadIdx.x;  // B*128
  if (idx >= ERD_B * 128) return;
  const int r = idx >> 7;
  const int cidx = idx & 127;
  prev[idx] = (cidx < ERD_DOUT) ? (_Float16)p[(size_t)r * ERD_DOUT + cidx]
                                : (_Float16)0.f;
}

// ---------------------------------------------------------------------------
// Host side
// ---------------------------------------------------------------------------
static void erd_launch_gemm(int mode,
    const _Float16* A, const _Float16* W, const float* bias,
    float* Cf, _Float16* Ch, float* out2, int out2_cols, int out2_stride,
    int M, int N, int K, hipStream_t s)
{
  dim3 grid(N / 128, M / 64), block(256);
  switch (mode) {
    case 0: hipLaunchKernelGGL((erd_gemm_wmma<0>), grid, block, 0, s,
                A, W, bias, Cf, Ch, out2, out2_cols, out2_stride, N, K); break;
    case 1: hipLaunchKernelGGL((erd_gemm_wmma<1>), grid, block, 0, s,
                A, W, bias, Cf, Ch, out2, out2_cols, out2_stride, N, K); break;
    case 2: hipLaunchKernelGGL((erd_gemm_wmma<2>), grid, block, 0, s,
                A, W, bias, Cf, Ch, out2, out2_cols, out2_stride, N, K); break;
    default: hipLaunchKernelGGL((erd_gemm_wmma<3>), grid, block, 0, s,
                A, W, bias, Cf, Ch, out2, out2_cols, out2_stride, N, K); break;
  }
}

extern "C" void kernel_launch(void* const* d_in, const int* in_sizes, int n_in,
                              void* d_out, int out_size, void* d_ws, size_t ws_size,
                              hipStream_t stream) {
  (void)in_sizes; (void)n_in; (void)out_size; (void)ws_size;

  // ---- inputs (setup_inputs dict order); encoder inputs are dead code ----
  const float* p         = (const float*)d_in[1];
  const float* fcin1_W   = (const float*)d_in[12];
  const float* fcin1_b   = (const float*)d_in[13];
  const float* fcin2_W   = (const float*)d_in[14];
  const float* fcin2_b   = (const float*)d_in[15];
  const float* Wih0      = (const float*)d_in[16];
  const float* Whh0      = (const float*)d_in[17];
  const float* bih0      = (const float*)d_in[18];
  const float* bhh0      = (const float*)d_in[19];
  const float* Wih1      = (const float*)d_in[20];
  const float* Whh1      = (const float*)d_in[21];
  const float* bih1      = (const float*)d_in[22];
  const float* bhh1      = (const float*)d_in[23];
  const float* fc1_W     = (const float*)d_in[24];
  const float* fc1_b     = (const float*)d_in[25];
  const float* fc2_W     = (const float*)d_in[26];
  const float* fc2_b     = (const float*)d_in[27];
  const float* fcout_W   = (const float*)d_in[28];
  const float* fcout_b   = (const float*)d_in[29];
  float* out = (float*)d_out;

  // ---- workspace carve-up (all padded dims multiples of 64) ----
  char* ws = (char*)d_ws;
  size_t off = 0;
  auto take = [&](size_t bytes) -> char* {
    char* ptr = ws + off;
    off = (off + bytes + 255) & ~(size_t)255;
    return ptr;
  };
  // f16 padded weights
  _Float16* wFcin1 = (_Float16*)take((size_t)512  * 128  * 2);
  _Float16* wFcin2 = (_Float16*)take((size_t)512  * 512  * 2);
  _Float16* wIh0   = (_Float16*)take((size_t)4096 * 512  * 2);
  _Float16* wHh0   = (_Float16*)take((size_t)4096 * 1024 * 2);
  _Float16* wIh1   = (_Float16*)take((size_t)4096 * 1024 * 2);
  _Float16* wHh1   = (_Float16*)take((size_t)4096 * 1024 * 2);
  _Float16* wFc1   = (_Float16*)take((size_t)512  * 1024 * 2);
  _Float16* wFc2   = (_Float16*)take((size_t)128  * 512  * 2);
  _Float16* wFco   = (_Float16*)take((size_t)128  * 128  * 2);
  // padded biases (f32); LSTM biases pre-summed (bih + bhh)
  float* bFcin1 = (float*)take(512  * 4);
  float* bFcin2 = (float*)take(512  * 4);
  float* bL0    = (float*)take(4096 * 4);
  float* bL1    = (float*)take(4096 * 4);
  float* bFc1   = (float*)take(512  * 4);
  float* bFc2   = (float*)take(128  * 4);
  float* bFco   = (float*)take(128  * 4);
  // activations / state
  _Float16* prev = (_Float16*)take((size_t)ERD_B * 128  * 2);
  _Float16* u1   = (_Float16*)take((size_t)ERD_B * 512  * 2);
  _Float16* u2   = (_Float16*)take((size_t)ERD_B * 512  * 2);
  _Float16* o1   = (_Float16*)take((size_t)ERD_B * 512  * 2);
  _Float16* h0   = (_Float16*)take((size_t)ERD_B * 1024 * 2);
  _Float16* h1   = (_Float16*)take((size_t)ERD_B * 1024 * 2);
  float*    c0   = (float*)   take((size_t)ERD_B * 1024 * 4);
  float*    c1   = (float*)   take((size_t)ERD_B * 1024 * 4);
  float*    gbuf = (float*)   take((size_t)ERD_B * 4096 * 4);

  // ---- setup: convert/pad weights, biases, initial state ----
  auto cvtW = [&](const float* src, _Float16* dst, int N, int K, int Np, int Kp) {
    int n = Np * Kp;
    hipLaunchKernelGGL(erd_cvt_weight_pad, dim3((n + 255) / 256), dim3(256), 0,
                       stream, src, dst, N, K, Np, Kp);
  };
  auto padB = [&](const float* b1, const float* b2, float* dst, int N, int Np) {
    hipLaunchKernelGGL(erd_pad_bias, dim3((Np + 255) / 256), dim3(256), 0,
                       stream, b1, b2, dst, N, Np);
  };
  cvtW(fcin1_W, wFcin1, 500,  96,   512,  128);
  cvtW(fcin2_W, wFcin2, 500,  500,  512,  512);
  cvtW(Wih0,    wIh0,   4000, 500,  4096, 512);
  cvtW(Whh0,    wHh0,   4000, 1000, 4096, 1024);
  cvtW(Wih1,    wIh1,   4000, 1000, 4096, 1024);
  cvtW(Whh1,    wHh1,   4000, 1000, 4096, 1024);
  cvtW(fc1_W,   wFc1,   500,  1000, 512,  1024);
  cvtW(fc2_W,   wFc2,   100,  500,  128,  512);
  cvtW(fcout_W, wFco,   96,   100,  128,  128);
  padB(fcin1_b, nullptr, bFcin1, 500,  512);
  padB(fcin2_b, nullptr, bFcin2, 500,  512);
  padB(bih0,    bhh0,    bL0,    4000, 4096);
  padB(bih1,    bhh1,    bL1,    4000, 4096);
  padB(fc1_b,   nullptr, bFc1,   500,  512);
  padB(fc2_b,   nullptr, bFc2,   100,  128);
  padB(fcout_b, nullptr, bFco,   96,   128);
  hipLaunchKernelGGL(erd_cvt_p, dim3((ERD_B * 128 + 255) / 256), dim3(256), 0,
                     stream, p, prev);
  // zero initial LSTM state (capture-safe async memset)
  hipMemsetAsync(h0, 0, (size_t)ERD_B * 1024 * 2, stream);
  hipMemsetAsync(h1, 0, (size_t)ERD_B * 1024 * 2, stream);
  hipMemsetAsync(c0, 0, (size_t)ERD_B * 1024 * 4, stream);
  hipMemsetAsync(c1, 0, (size_t)ERD_B * 1024 * 4, stream);

  // ---- 25 autoregressive decoder steps ----
  const dim3 pwGrid(ERD_B * 1024 / 256), pwBlk(256);
  for (int t = 0; t < ERD_TOUT; ++t) {
    // u1 = relu(prev @ fcin1^T + b)
    erd_launch_gemm(2, prev, wFcin1, bFcin1, nullptr, u1, nullptr, 0, 0,
                    ERD_B, 512, 128, stream);
    // u2 = relu(u1 @ fcin2^T + b)
    erd_launch_gemm(2, u1, wFcin2, bFcin2, nullptr, u2, nullptr, 0, 0,
                    ERD_B, 512, 512, stream);
    // gates = u2 @ Wih0^T + (bih0+bhh0);  gates += h0 @ Whh0^T
    erd_launch_gemm(0, u2, wIh0, bL0, gbuf, nullptr, nullptr, 0, 0,
                    ERD_B, 4096, 512, stream);
    erd_launch_gemm(1, h0, wHh0, nullptr, gbuf, nullptr, nullptr, 0, 0,
                    ERD_B, 4096, 1024, stream);
    hipLaunchKernelGGL(erd_lstm_pointwise, pwGrid, pwBlk, 0, stream,
                       gbuf, c0, h0);
    // gates = h0 @ Wih1^T + (bih1+bhh1);  gates += h1 @ Whh1^T
    erd_launch_gemm(0, h0, wIh1, bL1, gbuf, nullptr, nullptr, 0, 0,
                    ERD_B, 4096, 1024, stream);
    erd_launch_gemm(1, h1, wHh1, nullptr, gbuf, nullptr, nullptr, 0, 0,
                    ERD_B, 4096, 1024, stream);
    hipLaunchKernelGGL(erd_lstm_pointwise, pwGrid, pwBlk, 0, stream,
                       gbuf, c1, h1);
    // o1 = relu(h1 @ fc1^T + b) ; o2 = relu(o1 @ fc2^T + b)
    erd_launch_gemm(2, h1, wFc1, bFc1, nullptr, o1, nullptr, 0, 0,
                    ERD_B, 512, 1024, stream);
    erd_launch_gemm(2, o1, wFc2, bFc2, nullptr, u1 /*reuse as o2*/, nullptr, 0, 0,
                    ERD_B, 128, 512, stream);
    // prev = relu(o2 @ fcout^T + b); also emit f32 slice out[:, t, :96]
    erd_launch_gemm(3, u1, wFco, bFco, nullptr, prev,
                    out + (size_t)t * ERD_DOUT, ERD_DOUT, ERD_TOUT * ERD_DOUT,
                    ERD_B, 128, 128, stream);
  }
}
